// SelfAttention_84421877170208
// MI455X (gfx1250) — matrix operand order repaired and checked by reading, not compiled, not run on hardware
//
#include <hip/hip_runtime.h>
#include <hip/hip_bf16.h>

#define D      256          // FEATURE_DIM
#define H      8            // HEADS
#define DH     2048         // D*H
#define NTOK   16384        // b*n = 4*4096
#define TILE_M 16

typedef __attribute__((ext_vector_type(16))) __bf16 v16bf;
typedef __attribute__((ext_vector_type(8)))  __bf16 v8bf;
typedef __attribute__((ext_vector_type(8)))  float  v8f;

union AFrag { v16bf v; v8bf h[2]; };

// workspace layout (bf16, all transposed to N-major so B-frags are contiguous):
//   WqT[2048][256], WkT[2048][256], WvT[2048][256], WoT[256][2048]
#define WQT_OFF 0
#define WKT_OFF (2048 * 256)
#define WVT_OFF (2 * 2048 * 256)
#define WOT_OFF (3 * 2048 * 256)
#define WTOT    (4 * 2048 * 256)

__global__ void convert_weights_bf16T(const float* __restrict__ Wq,
                                      const float* __restrict__ Wk,
                                      const float* __restrict__ Wv,
                                      const float* __restrict__ Wo,
                                      __bf16* __restrict__ wsbf) {
  int idx = blockIdx.x * blockDim.x + threadIdx.x;
  if (idx >= WTOT) return;
  int m = idx / (2048 * 256);
  int r = idx % (2048 * 256);
  float v;
  if (m == 0)      { int n = r / 256,  k = r % 256;  v = Wq[k * 2048 + n]; }
  else if (m == 1) { int n = r / 256,  k = r % 256;  v = Wk[k * 2048 + n]; }
  else if (m == 2) { int n = r / 256,  k = r % 256;  v = Wv[k * 2048 + n]; }
  else             { int n = r / 2048, k = r % 2048; v = Wo[k * 256  + n]; }
  wsbf[idx] = (__bf16)v;
}

// One projection GEMM: (16 x 256) A-tile (in registers) times WT (N-major
// 2048 x 256 bf16), bias added, bf16 result tile (16 x 2048) into LDS.
// Fixed pointers per call -> no per-iteration select chains.
__device__ __forceinline__ void proj_gemm(const AFrag a[8],
                                          const __bf16* __restrict__ Wbase,
                                          const float* __restrict__ bias,
                                          __bf16* __restrict__ dst,
                                          int wave, int lcol, int half) {
  for (int nt = wave; nt < DH / 16; nt += 8) {
    int n0 = nt << 4;
    const __bf16* W = Wbase + (size_t)(n0 + lcol) * D;
    __builtin_prefetch(W + 128 * D, 0, 1);   // this wave's next n-tile column
    v8f c = {};
    #pragma unroll
    for (int kk = 0; kk < 8; ++kk) {
      v16bf bf = *(const v16bf*)(W + kk * 32 + half * 16);
      c = __builtin_amdgcn_wmma_f32_16x16x32_bf16(false, a[kk].v, false, bf,
                                                  (short)0, c, false, false);
    }
    float bb = bias[n0 + lcol];
    #pragma unroll
    for (int j = 0; j < 8; ++j)
      dst[(size_t)(j + half * 8) * DH + n0 + lcol] = (__bf16)(c[j] + bb);
  }
}

__global__ __launch_bounds__(256, 1)
void fused_attn(const float* __restrict__ x,
                const float* __restrict__ bq,
                const float* __restrict__ bk,
                const float* __restrict__ bv,
                const float* __restrict__ bo,
                const __bf16* __restrict__ wsbf,
                float* __restrict__ out) {
  extern __shared__ char lds[];
  __bf16* xA   = (__bf16*)lds;                       // 16 x 256       (8 KB)
  __bf16* Qs   = (__bf16*)(lds + TILE_M * D * 2);    // 16 x 2048      (64 KB)
  __bf16* Ks   = Qs + TILE_M * DH;                   // 16 x 2048      (64 KB)
  __bf16* Vs   = Ks + TILE_M * DH;                   // 16 x 2048      (64 KB)
  __bf16* Outs = Vs + TILE_M * DH;                   // 16 x 2048      (64 KB)
  float*  P    = (float*)(Outs + TILE_M * DH);       // 16 x 8 x 8     (4 KB)

  const int tid   = threadIdx.x;
  const int wave  = tid >> 5;
  const int lane  = tid & 31;
  const int half  = lane >> 4;       // which 16-lane half of the wave
  const int lcol  = lane & 15;       // column (B/C/D) / row (A) index
  const int tile0 = blockIdx.x * TILE_M;

  // ---- phase 1: stage x tile (16 x 256) into LDS as bf16 ----
  for (int i = tid; i < TILE_M * D; i += 256) {
    int r = i >> 8, c = i & 255;
    xA[i] = (__bf16)x[(size_t)(tile0 + r) * D + c];
  }
  __syncthreads();

  // preload the 8 A-fragments (K = 256 = 8 * 32) of the x tile into registers
  AFrag a[8];
  {
    const __bf16* row = xA + lcol * D;
    #pragma unroll
    for (int kk = 0; kk < 8; ++kk) {
      a[kk].h[0] = *(const v8bf*)(row + kk * 32 + half * 16);
      a[kk].h[1] = *(const v8bf*)(row + kk * 32 + half * 16 + 8);
    }
  }

  // ---- phase 2: Q/K/V projections (three fixed-pointer GEMM passes) ----
  proj_gemm(a, wsbf + WQT_OFF, bq, Qs, wave, lcol, half);
  proj_gemm(a, wsbf + WKT_OFF, bk, Ks, wave, lcol, half);
  proj_gemm(a, wsbf + WVT_OFF, bv, Vs, wave, lcol, half);
  __syncthreads();

  // ---- phase 3: scores via WMMA; each wave handles token pair (2w, 2w+1) ----
  // 16x16 tile rows/cols = {tok0.h0..h7, tok1.h0..h7}; keep the two diagonal 8x8 blocks
  {
    const int ptok  = (lcol >> 3);         // 0/1 within the pair
    const int phead = lcol & 7;
    const __bf16* qrow = Qs + (size_t)(2 * wave + ptok) * DH + phead * D;
    const __bf16* kcol = Ks + (size_t)(2 * wave + ptok) * DH + phead * D;
    v8f c = {};
    #pragma unroll
    for (int kk = 0; kk < 8; ++kk) {
      AFrag qa;
      qa.h[0] = *(const v8bf*)(qrow + kk * 32 + half * 16);
      qa.h[1] = *(const v8bf*)(qrow + kk * 32 + half * 16 + 8);
      v16bf kb = *(const v16bf*)(kcol + kk * 32 + half * 16);
      c = __builtin_amdgcn_wmma_f32_16x16x32_bf16(false, qa.v, false, kb,
                                                  (short)0, c, false, false);
    }
    const float scale = 0.0625f;           // 1/sqrt(256)
    #pragma unroll
    for (int j = 0; j < 8; ++j) {
      int row = j + half * 8;
      if ((row >> 3) == (lcol >> 3)) {     // same-token diagonal block only
        int tk = 2 * wave + (row >> 3);
        P[tk * 64 + (row & 7) * 8 + (lcol & 7)] = c[j] * scale;
      }
    }
  }
  __syncthreads();

  // ---- softmax over g (8 heads) for each of the 128 (token, head) rows ----
  if (tid < TILE_M * H) {
    float* row = P + tid * 8;
    float mx = row[0];
    #pragma unroll
    for (int g = 1; g < 8; ++g) mx = fmaxf(mx, row[g]);
    float e[8], s = 0.f;
    #pragma unroll
    for (int g = 0; g < 8; ++g) { e[g] = __expf(row[g] - mx); s += e[g]; }
    float inv = 1.f / s;
    #pragma unroll
    for (int g = 0; g < 8; ++g) row[g] = e[g] * inv;
  }
  __syncthreads();

  // ---- phase 4: out = attn @ V; 8 consecutive d per thread-task so V reads
  //      are 16B vector LDS loads and P reads are amortized 8x ----
  for (int i = tid; i < TILE_M * DH / 8; i += 256) {   // 4096 vector tasks
    int tk = i >> 8;                 // 256 eight-wide chunks per token
    int c  = i & 255;
    int hh = c >> 5;                 // 32 chunks per head
    int d8 = (c & 31) << 3;          // d base within the head
    const float*  pr = P  + tk * 64 + hh * 8;
    const __bf16* vb = Vs + (size_t)tk * DH + d8;
    float acc[8] = {0.f, 0.f, 0.f, 0.f, 0.f, 0.f, 0.f, 0.f};
    #pragma unroll
    for (int g = 0; g < 8; ++g) {
      v8bf vv = *(const v8bf*)(vb + g * D);
      float pg = pr[g];
      #pragma unroll
      for (int e = 0; e < 8; ++e) acc[e] += pg * (float)vv[e];
    }
    v8bf o;
    #pragma unroll
    for (int e = 0; e < 8; ++e) o[e] = (__bf16)acc[e];
    *(v8bf*)(Outs + (size_t)tk * DH + hh * D + d8) = o;
  }
  __syncthreads();

  // ---- phase 5: final projection Out(16x2048) @ Wo(2048x256) + bo ----
  for (int nt = wave; nt < D / 16; nt += 8) {
    int n0 = nt << 4;
    const __bf16* W    = wsbf + WOT_OFF + (size_t)(n0 + lcol) * DH;
    const __bf16* orow = Outs + lcol * DH;
    v8f c = {};
    #pragma unroll 8
    for (int kk = 0; kk < DH / 32; ++kk) {
      AFrag oa;
      oa.h[0] = *(const v8bf*)(orow + kk * 32 + half * 16);
      oa.h[1] = *(const v8bf*)(orow + kk * 32 + half * 16 + 8);
      v16bf wb = *(const v16bf*)(W + kk * 32 + half * 16);
      c = __builtin_amdgcn_wmma_f32_16x16x32_bf16(false, oa.v, false, wb,
                                                  (short)0, c, false, false);
    }
    float bb = bo[n0 + lcol];
    #pragma unroll
    for (int j = 0; j < 8; ++j)
      out[(size_t)(tile0 + j + half * 8) * D + n0 + lcol] = c[j] + bb;
  }
}

extern "C" void kernel_launch(void* const* d_in, const int* in_sizes, int n_in,
                              void* d_out, int out_size, void* d_ws, size_t ws_size,
                              hipStream_t stream) {
  const float* x  = (const float*)d_in[0];
  const float* Wq = (const float*)d_in[1];
  const float* bq = (const float*)d_in[2];
  const float* Wk = (const float*)d_in[3];
  const float* bk = (const float*)d_in[4];
  const float* Wv = (const float*)d_in[5];
  const float* bv = (const float*)d_in[6];
  const float* Wo = (const float*)d_in[7];
  const float* bo = (const float*)d_in[8];
  __bf16* wsbf = (__bf16*)d_ws;
  float*  out  = (float*)d_out;

  convert_weights_bf16T<<<(WTOT + 255) / 256, 256, 0, stream>>>(Wq, Wk, Wv, Wo, wsbf);

  // LDS: xA (8K) + Q/K/V/Out tiles (4 * 64K) + probs (4K) = 274432 B (< 320 KB WGP)
  size_t lds_bytes = (size_t)TILE_M * D * 2 + 4 * (size_t)TILE_M * DH * 2 + TILE_M * 64 * 4;
  fused_attn<<<NTOK / TILE_M, 256, lds_bytes, stream>>>(x, bq, bk, bv, bo, wsbf, out);
}
